// TransformerBlock_28664611734079
// MI455X (gfx1250) — matrix operand order repaired and checked by reading, not compiled, not run on hardware
//
#include <hip/hip_runtime.h>
#include <cstdint>
#include <cstddef>

#define Bsz  4
#define Tseq 2048
#define Dm   768
#define Hh   12
#define DHd  64
#define DFFd 3072
#define Mtot (Bsz * Tseq)

typedef __attribute__((ext_vector_type(16))) __bf16 v16bf;
typedef __attribute__((ext_vector_type(8)))  float  v8f;
typedef __attribute__((ext_vector_type(4)))  unsigned int v4u;
typedef __attribute__((ext_vector_type(8)))  unsigned int v8u;

__device__ __forceinline__ v16bf pack16(uint4 lo, uint4 hi) {
    union { uint4 u[2]; v16bf v; } t;
    t.u[0] = lo; t.u[1] = hi;
    return t.v;
}

__device__ __forceinline__ v8f wmma_bf16(v16bf a, v16bf b, v8f c) {
    // (neg_a, A, neg_b, B, c_mod, C, reuse_a, reuse_b)
    return __builtin_amdgcn_wmma_f32_16x16x32_bf16(false, a, false, b, (short)0, c, false, false);
}

__device__ __forceinline__ v8f vzero8() {
    v8f z;
#pragma unroll
    for (int i = 0; i < 8; i++) z[i] = 0.0f;
    return z;
}

// ---------------------------------------------------------------------------
// Tensor Data Mover: DMA a 2-D bf16 tile Global -> LDS (per-wave issue).
// D# layout per CDNA5 ISA (groups 2/3 zero: 2-D tile).
// tile_w elems/row (row contiguous), tile_h rows, row_stride in elements.
// ---------------------------------------------------------------------------
__device__ __forceinline__ void tdm_load_2d(const void* gptr, unsigned lds_off,
                                            unsigned tile_w, unsigned tile_h,
                                            unsigned long long row_stride) {
    const unsigned long long ga = (unsigned long long)gptr;
    v4u g0;
    g0[0] = 1u;                                        // count=1 (valid), user mode
    g0[1] = lds_off;                                   // lds_addr (bytes)
    g0[2] = (unsigned)(ga & 0xffffffffu);              // global_addr[31:0]
    g0[3] = (unsigned)((ga >> 32) & 0x01ffffffu)       // global_addr[56:32]
          | (2u << 30);                                // type = 2 ("image")
    v8u g1;
    g1[0] = (1u << 16);                                // data_size=1 -> 2 bytes
    g1[1] = (tile_w & 0xffffu) << 16;                  // tensor_dim0[15:0]
    g1[2] = (tile_w >> 16) | ((tile_h & 0xffffu) << 16); // dim0[31:16] | dim1[15:0]
    g1[3] = (tile_h >> 16) | (tile_w << 16);           // dim1[31:16] | tile_dim0
    g1[4] = tile_h;                                    // tile_dim1 (tile_dim2=0)
    g1[5] = (unsigned)(row_stride & 0xffffffffull);    // tensor_dim0_stride[31:0]
    g1[6] = (unsigned)((row_stride >> 32) & 0xffffull);// stride[47:32] | dim1_stride lo
    g1[7] = 0u;
    v4u gz;
    gz[0] = 0u; gz[1] = 0u; gz[2] = 0u; gz[3] = 0u;
    asm volatile("tensor_load_to_lds %0, %1, %2, %3"
                 :: "s"(g0), "s"(g1), "s"(gz), "s"(gz)
                 : "memory");
}

__device__ __forceinline__ void tdm_wait() {
#if __has_builtin(__builtin_amdgcn_s_wait_tensorcnt)
    __builtin_amdgcn_s_wait_tensorcnt(0);
#else
    asm volatile("s_wait_tensorcnt 0x0" ::: "memory");
#endif
}

// generic pointer -> LDS byte offset (aperture: offset is the low 32 bits of the address)
__device__ __forceinline__ unsigned lds_off(const void* p) {
    return (unsigned)(uintptr_t)p;
}

// ---------------------------------------------------------------------------
// f32 -> bf16 weight pack
// ---------------------------------------------------------------------------
__global__ __launch_bounds__(256) void cvt_kernel(const float* __restrict__ s,
                                                  __bf16* __restrict__ d, int n) {
    int i0 = blockIdx.x * 1024 + threadIdx.x;
#pragma unroll
    for (int j = 0; j < 4; j++) {
        int i = i0 + j * 256;
        if (i < n) d[i] = (__bf16)s[i];
    }
}

// ---------------------------------------------------------------------------
// LayerNorm: one block per row (D=768, 256 threads x 3 elems), bf16 output
// ---------------------------------------------------------------------------
__global__ __launch_bounds__(256) void ln_kernel(const float* __restrict__ x,
                                                 const float* __restrict__ g,
                                                 const float* __restrict__ b,
                                                 __bf16* __restrict__ out) {
    const int row = blockIdx.x;
    const int t   = threadIdx.x;
    const float* xr = x + (size_t)row * Dm;
    float v0 = xr[t], v1 = xr[t + 256], v2 = xr[t + 512];
    float s  = v0 + v1 + v2;
    float s2 = v0 * v0 + v1 * v1 + v2 * v2;
#pragma unroll
    for (int off = 16; off >= 1; off >>= 1) {
        s  += __shfl_xor(s,  off, 32);
        s2 += __shfl_xor(s2, off, 32);
    }
    __shared__ float red[2][8];
    const int wave = t >> 5, lane = t & 31;
    if (lane == 0) { red[0][wave] = s; red[1][wave] = s2; }
    __syncthreads();
    float ts = 0.f, ts2 = 0.f;
#pragma unroll
    for (int i = 0; i < 8; i++) { ts += red[0][i]; ts2 += red[1][i]; }
    const float mu   = ts * (1.0f / Dm);
    const float var  = ts2 * (1.0f / Dm) - mu * mu;
    const float rstd = rsqrtf(var + 1e-5f);
    __bf16* orow = out + (size_t)row * Dm;
    orow[t]       = (__bf16)((v0 - mu) * rstd * g[t]       + b[t]);
    orow[t + 256] = (__bf16)((v1 - mu) * rstd * g[t + 256] + b[t + 256]);
    orow[t + 512] = (__bf16)((v2 - mu) * rstd * g[t + 512] + b[t + 512]);
}

// ---------------------------------------------------------------------------
// WMMA GEMM: C[M,N] = A[M,K] @ W[K,N] + bias (+gelu | +residual)
// EPI 0: bias, bf16 out.  EPI 1: bias+gelu, bf16 out.  EPI 2: bias+resid, f32 out.
// Block tile 128x128, BK=32, 8 waves (4x2), wave tile 32x64 (2x4 WMMA tiles).
// A tile staged by TDM (per-wave 16-row slice); W tile transposed by VALU.
// ---------------------------------------------------------------------------
template <int EPI>
__global__ __launch_bounds__(256) void gemm_kernel(const __bf16* __restrict__ A,
                                                   const __bf16* __restrict__ W,
                                                   const float* __restrict__ bias,
                                                   const float* __restrict__ resid,
                                                   void* __restrict__ outp,
                                                   int M, int N, int K) {
    __shared__ __bf16 As[128][32];  // [m][k]
    __shared__ __bf16 Bt[128][32];  // transposed: [n][k]
    const int tid  = threadIdx.x;
    const int wave = tid >> 5, lane = tid & 31;
    const int waveM = wave >> 1, waveN = wave & 1;
    const int mrow  = lane & 15;
    const int half8  = (lane < 16) ? 0 : 8;
    const int half16 = (lane < 16) ? 0 : 16;
    const int bm = blockIdx.y, bn = blockIdx.x;

    v8f acc[2][4];
#pragma unroll
    for (int i = 0; i < 2; i++)
#pragma unroll
        for (int j = 0; j < 4; j++) acc[i][j] = vzero8();

    const int bkrow = tid >> 3, bnoff = (tid & 7) * 16;    // W: 8 threads/row
    const unsigned ldsA = lds_off(&As[0][0]) + (unsigned)wave * 16u * 32u * 2u;

    for (int k0 = 0; k0 < K; k0 += 32) {
        // TDM: this wave's 16 rows of the A tile, Global -> LDS
        tdm_load_2d(A + (size_t)(bm * 128 + wave * 16) * K + k0, ldsA,
                    32u, 16u, (unsigned long long)K);
        {   // stage W tile transposed (overlaps the TDM)
            const __bf16* wp = W + (size_t)(k0 + bkrow) * N + bn * 128 + bnoff;
#pragma unroll
            for (int j = 0; j < 16; j++) Bt[bnoff + j][bkrow] = wp[j];
        }
        if (k0 + 32 < K)
            __builtin_prefetch(W + (size_t)(k0 + 32 + bkrow) * N + bn * 128 + bnoff, 0, 1);
        tdm_wait();
        __syncthreads();

        v16bf af[2], bf[4];
#pragma unroll
        for (int mt = 0; mt < 2; mt++) {
            const __bf16* p = &As[waveM * 32 + mt * 16 + mrow][0];
            af[mt] = pack16(*(const uint4*)(p + half8), *(const uint4*)(p + 16 + half8));
        }
#pragma unroll
        for (int nt = 0; nt < 4; nt++) {
            const __bf16* p = &Bt[waveN * 64 + nt * 16 + mrow][0];
            bf[nt] = pack16(*(const uint4*)(p + half8), *(const uint4*)(p + 16 + half8));
        }
#pragma unroll
        for (int mt = 0; mt < 2; mt++)
#pragma unroll
            for (int nt = 0; nt < 4; nt++)
                acc[mt][nt] = wmma_bf16(af[mt], bf[nt], acc[mt][nt]);
        __syncthreads();
    }

#pragma unroll
    for (int mt = 0; mt < 2; mt++)
#pragma unroll
        for (int nt = 0; nt < 4; nt++) {
            const int col = bn * 128 + waveN * 64 + nt * 16 + mrow;
            const float bv = bias[col];
#pragma unroll
            for (int r = 0; r < 8; r++) {
                const int row = bm * 128 + waveM * 32 + mt * 16 + half8 + r;
                float v = acc[mt][nt][r] + bv;
                if constexpr (EPI == 1)
                    v = 0.5f * v * (1.0f + erff(v * 0.70710678118654752f));
                if constexpr (EPI == 2) {
                    v += resid[(size_t)row * N + col];
                    ((float*)outp)[(size_t)row * N + col] = v;
                } else {
                    ((__bf16*)outp)[(size_t)row * N + col] = (__bf16)v;
                }
            }
        }
}

// ---------------------------------------------------------------------------
// Flash attention: one block per (qt, h, b). 8 waves; wave owns 16 q rows.
// qkv: [B*T, 3*D] bf16 (Q | K | V).  out: [B*T, D] bf16.
// K tile staged by TDM; V tile transposed by VALU; causal mask only on the
// diagonal k-tile.
// ---------------------------------------------------------------------------
__global__ __launch_bounds__(256) void attn_kernel(const __bf16* __restrict__ qkv,
                                                   __bf16* __restrict__ out) {
    __shared__ __bf16 Ks[128][64];       // K tile [key][dh]
    __shared__ __bf16 Vt[64][128];       // V tile transposed [dh][key]
    __shared__ __bf16 Ps[8][16][128];    // per-wave P strip (C-layout -> A-layout)
    const int qt = blockIdx.x, h = blockIdx.y, b = blockIdx.z;
    const int tid = threadIdx.x, wave = tid >> 5, lane = tid & 31;
    const int mrow = lane & 15;
    const int half8  = (lane < 16) ? 0 : 8;
    const int half16 = (lane < 16) ? 0 : 16;
    const size_t ldq = 3 * Dm;

    // Q fragments, resident for whole loop (loaded from global in AMD k order)
    v16bf qf[2];
    {
        const __bf16* qp = qkv + (size_t)(b * Tseq + qt * 128 + wave * 16 + mrow) * ldq + h * DHd;
#pragma unroll
        for (int c = 0; c < 2; c++) {
            const __bf16* p = qp + c * 32;
            qf[c] = pack16(*(const uint4*)(p + half8), *(const uint4*)(p + 16 + half8));
        }
    }

    v8f Oacc[4];
#pragma unroll
    for (int vt = 0; vt < 4; vt++) Oacc[vt] = vzero8();
    float mrun[8], lrun[8];
#pragma unroll
    for (int r = 0; r < 8; r++) { mrun[r] = -3.0e38f; lrun[r] = 0.0f; }
    const float scale = 0.125f;  // 1/sqrt(64)
    const int ldr = tid >> 2, ldo = (tid & 3) * 16;
    const unsigned ldsK = lds_off(&Ks[0][0]) + (unsigned)wave * 16u * 64u * 2u;

    for (int kt = 0; kt <= qt; kt++) {
        // TDM: this wave's 16 key rows, Global -> LDS
        tdm_load_2d(qkv + (size_t)(b * Tseq + kt * 128 + wave * 16) * ldq + Dm + h * DHd,
                    ldsK, 64u, 16u, (unsigned long long)ldq);
        {   // stage V transposed (overlaps the TDM)
            const __bf16* vp = qkv + (size_t)(b * Tseq + kt * 128 + ldr) * ldq + 2 * Dm + h * DHd + ldo;
#pragma unroll
            for (int j = 0; j < 16; j++) Vt[ldo + j][ldr] = vp[j];
        }
        tdm_wait();
        __syncthreads();

        // S strip: 16 q-rows x 128 keys = 8 WMMA tiles
        const bool diag = (kt == qt);
        float sreg[8][8];
#pragma unroll
        for (int nt = 0; nt < 8; nt++) {
            v8f s = vzero8();
#pragma unroll
            for (int c = 0; c < 2; c++) {
                const __bf16* p = &Ks[nt * 16 + mrow][c * 32 + half8];
                v16bf kf = pack16(*(const uint4*)p, *(const uint4*)(p + 16));
                s = wmma_bf16(qf[c], kf, s);
            }
            if (!diag) {
#pragma unroll
                for (int r = 0; r < 8; r++) sreg[nt][r] = s[r] * scale;
            } else {
                const int key = nt * 16 + mrow;
#pragma unroll
                for (int r = 0; r < 8; r++) {
                    const int qi = wave * 16 + half8 + r;
                    sreg[nt][r] = (key <= qi) ? s[r] * scale : -3.0e38f;
                }
            }
        }

        // online softmax (row reductions across the 16-lane half)
#pragma unroll
        for (int r = 0; r < 8; r++) {
            float mx = sreg[0][r];
#pragma unroll
            for (int nt = 1; nt < 8; nt++) mx = fmaxf(mx, sreg[nt][r]);
#pragma unroll
            for (int off = 1; off < 16; off <<= 1) mx = fmaxf(mx, __shfl_xor(mx, off, 32));
            const float mnew = fmaxf(mrun[r], mx);
            const float corr = __expf(mrun[r] - mnew);
            mrun[r] = mnew;
            float lsum = 0.0f;
#pragma unroll
            for (int nt = 0; nt < 8; nt++) {
                const float p = __expf(sreg[nt][r] - mnew);
                Ps[wave][half8 + r][nt * 16 + mrow] = (__bf16)p;
                lsum += p;
            }
#pragma unroll
            for (int off = 1; off < 16; off <<= 1) lsum += __shfl_xor(lsum, off, 32);
            lrun[r] = lrun[r] * corr + lsum;
#pragma unroll
            for (int vt = 0; vt < 4; vt++) Oacc[vt][r] = Oacc[vt][r] * corr;
        }
        __syncthreads();

        // O += P @ V  (4 key-chunks of 32, 4 dh tiles of 16)
#pragma unroll
        for (int c = 0; c < 4; c++) {
            const __bf16* pp = &Ps[wave][mrow][c * 32];
            v16bf pf = pack16(*(const uint4*)(pp + half8), *(const uint4*)(pp + 16 + half8));
#pragma unroll
            for (int vt = 0; vt < 4; vt++) {
                const __bf16* vp = &Vt[vt * 16 + mrow][c * 32 + half8];
                v16bf vf = pack16(*(const uint4*)vp, *(const uint4*)(vp + 16));
                Oacc[vt] = wmma_bf16(pf, vf, Oacc[vt]);
            }
        }
        __syncthreads();
    }

#pragma unroll
    for (int vt = 0; vt < 4; vt++) {
        const int col = h * DHd + vt * 16 + mrow;
#pragma unroll
        for (int r = 0; r < 8; r++) {
            const int row = b * Tseq + qt * 128 + wave * 16 + half8 + r;
            out[(size_t)row * Dm + col] = (__bf16)(Oacc[vt][r] / lrun[r]);
        }
    }
}

// ---------------------------------------------------------------------------
// Host launch
// ---------------------------------------------------------------------------
extern "C" void kernel_launch(void* const* d_in, const int* in_sizes, int n_in,
                              void* d_out, int out_size, void* d_ws, size_t ws_size,
                              hipStream_t stream) {
    const float* x      = (const float*)d_in[0];
    const float* ln1_g  = (const float*)d_in[1];
    const float* ln1_b  = (const float*)d_in[2];
    const float* w_qkv  = (const float*)d_in[3];
    const float* b_qkv  = (const float*)d_in[4];
    const float* w_proj = (const float*)d_in[5];
    const float* b_proj = (const float*)d_in[6];
    const float* ln2_g  = (const float*)d_in[7];
    const float* ln2_b  = (const float*)d_in[8];
    const float* w_ff1  = (const float*)d_in[9];
    const float* b_ff1  = (const float*)d_in[10];
    const float* w_ff2  = (const float*)d_in[11];
    const float* b_ff2  = (const float*)d_in[12];

    char* ws = (char*)d_ws;
    auto take = [&](size_t bytes) {
        char* p = ws;
        ws += (bytes + 255) & ~(size_t)255;
        return p;
    };
    __bf16* wqkv_h  = (__bf16*)take((size_t)Dm * 3 * Dm * 2);
    __bf16* wproj_h = (__bf16*)take((size_t)Dm * Dm * 2);
    __bf16* wff1_h  = (__bf16*)take((size_t)Dm * DFFd * 2);
    __bf16* wff2_h  = (__bf16*)take((size_t)DFFd * Dm * 2);
    __bf16* hbuf    = (__bf16*)take((size_t)Mtot * Dm * 2);
    __bf16* qkvb    = (__bf16*)take((size_t)Mtot * 3 * Dm * 2);
    __bf16* attb    = (__bf16*)take((size_t)Mtot * Dm * 2);
    float*  x2b     = (float*) take((size_t)Mtot * Dm * 4);
    __bf16* ffb     = (__bf16*)take((size_t)Mtot * DFFd * 2);

    auto cvt = [&](const float* s, __bf16* d, size_t n) {
        cvt_kernel<<<dim3((unsigned)((n + 1023) / 1024)), dim3(256), 0, stream>>>(s, d, (int)n);
    };
    cvt(w_qkv,  wqkv_h,  (size_t)Dm * 3 * Dm);
    cvt(w_proj, wproj_h, (size_t)Dm * Dm);
    cvt(w_ff1,  wff1_h,  (size_t)Dm * DFFd);
    cvt(w_ff2,  wff2_h,  (size_t)DFFd * Dm);

    // LN1 -> h
    ln_kernel<<<dim3(Mtot), dim3(256), 0, stream>>>(x, ln1_g, ln1_b, hbuf);
    // qkv = h @ w_qkv + b_qkv  (bf16 out)
    gemm_kernel<0><<<dim3(3 * Dm / 128, Mtot / 128), dim3(256), 0, stream>>>(
        hbuf, wqkv_h, b_qkv, nullptr, qkvb, Mtot, 3 * Dm, Dm);
    // causal flash attention
    attn_kernel<<<dim3(Tseq / 128, Hh, Bsz), dim3(256), 0, stream>>>(qkvb, attb);
    // x2 = x + att @ w_proj + b_proj  (f32 out)
    gemm_kernel<2><<<dim3(Dm / 128, Mtot / 128), dim3(256), 0, stream>>>(
        attb, wproj_h, b_proj, x, x2b, Mtot, Dm, Dm);
    // LN2 -> h (reuse)
    ln_kernel<<<dim3(Mtot), dim3(256), 0, stream>>>(x2b, ln2_g, ln2_b, hbuf);
    // ff = gelu(h @ w_ff1 + b_ff1)  (bf16 out)
    gemm_kernel<1><<<dim3(DFFd / 128, Mtot / 128), dim3(256), 0, stream>>>(
        hbuf, wff1_h, b_ff1, nullptr, ffb, Mtot, DFFd, Dm);
    // out = x2 + ff @ w_ff2 + b_ff2  (f32 out)
    gemm_kernel<2><<<dim3(Dm / 128, Mtot / 128), dim3(256), 0, stream>>>(
        ffb, wff2_h, b_ff2, x2b, (float*)d_out, Mtot, Dm, DFFd);
}
